// DenseGINEConv_11553462026952
// MI455X (gfx1250) — matrix operand order repaired and checked by reading, not compiled, not run on hardware
//
#include <hip/hip_runtime.h>

// ---------------------------------------------------------------------------
// DenseGINEConv for MI455X (gfx1250, wave32, WMMA bf16)
//   agg = 0
//   edge kernel:  msg = MLP1(bf16(x[src]+edge_attr));  atomicAdd(agg[dst], msg)
//   node kernel:  out = MLP2(bf16((1+eps)*x + agg))
// MLP = Linear(128,128) -> GELU -> Linear(128,128), f32 accumulate via
// v_wmma_f32_16x16x32_bf16, biases folded into the accumulator init.
// Each wave owns 32 rows (2 M-tiles) so every B fragment loaded from LDS
// feeds two WMMAs. GELU uses a branch-free tanh form (error << bf16 quant
// noise of the hidden layer) built on v_exp_f32 / v_rcp_f32.
// ---------------------------------------------------------------------------

#define DD 128            // feature dim
#define BR 128            // rows (edges / nodes) per workgroup
#define NTHREADS 128      // 4 waves of 32; each wave handles 32 rows
#define NNODES 50000
#define NEDGES 600000

typedef __bf16 bf16;
typedef __attribute__((ext_vector_type(16))) __bf16 v16bf;
typedef __attribute__((ext_vector_type(8)))  __bf16 v8bf;
typedef __attribute__((ext_vector_type(4)))  __bf16 v4bf;
typedef __attribute__((ext_vector_type(8)))  float  v8f;

// Branch-free GELU (tanh form). Max deviation from exact-erf GELU ~1e-3,
// below the bf16 quantization applied to the hidden activations anyway.
//   tanh(z) = 1 - 2/(exp2(2*log2e*z) + 1)   (saturates naturally, no branches)
__device__ __forceinline__ float gelu_fast(float v) {
  float z = 0.7978845608028654f * v * __builtin_fmaf(0.044715f * v, v, 1.0f);
  float e = __builtin_amdgcn_exp2f(z * 2.8853900817779268f);   // 2/ln(2)
  float t = __builtin_fmaf(-2.0f, __builtin_amdgcn_rcpf(e + 1.0f), 1.0f);
  return 0.5f * v * (1.0f + t);
}

// Stage a 128x128 f32 weight matrix, transposed, as bf16 into LDS:
//   sW[n*DD + k] = bf16(W[k*DD + n])
// Transposed layout makes each lane's WMMA B-fragment one contiguous 32-byte
// LDS read.
__device__ __forceinline__ void stage_w_t(const float* __restrict__ W,
                                          bf16* __restrict__ sW) {
  for (int i = threadIdx.x; i < DD * DD; i += NTHREADS) {
    int n = i >> 7;
    int k = i & (DD - 1);
    sW[i] = (bf16)W[k * DD + n];
  }
}

__device__ __forceinline__ v8f wmma_bf16(v16bf a, v16bf b, v8f c) {
  return __builtin_amdgcn_wmma_f32_16x16x32_bf16(
      /*neg_a=*/false, a, /*neg_b=*/false, b,
      /*c_mod=*/(short)0, c, /*reuse_a=*/false, /*reuse_b=*/false);
}

// One Linear layer for this wave's 2 M-tiles (rows rowBase..rowBase+31) across
// all 128 output columns.
// A-fragment (16-bit A 16x32): lane holds row (lane&15);
//   16 consecutive K starting at kof=(lane<16)?0:16 (matches B order).
// B-fragment (32x16): lane holds col (lane&15); 16 consecutive K starting at
//   (lane<16)?0:16 -> contiguous in the transposed sW.
// C/D: VGPR r holds rows (r, r+8) at col (lane&15).
// WB == 0: GELU -> bf16 -> sXw (own rows only, no cross-wave hazard)
// WB == 1: f32 atomic scatter-add to gout[dsts[..]*DD + n]
// WB == 2: f32 store to gout[(gRowBase+row)*DD + n]
template <int WB>
__device__ __forceinline__ void run_layer(const bf16* __restrict__ sXr,
                                          bf16* __restrict__ sXw,
                                          const bf16* __restrict__ sW,
                                          const float* __restrict__ bias,
                                          int lane, int rowBase, int rows,
                                          float* __restrict__ gout,
                                          const int* __restrict__ dsts,
                                          int gRowBase) {
  const int m     = lane & 15;
  const int kofA  = (lane & 16) ? 16 : 0;
  const int kofB  = (lane & 16) ? 16 : 0;
  const int rHalf = (lane & 16) ? 8 : 0;

  v16bf A[2][4];
#pragma unroll
  for (int t = 0; t < 2; ++t) {
    const bf16* ar = sXr + (rowBase + 16 * t + m) * DD + kofA;
#pragma unroll
    for (int ks = 0; ks < 4; ++ks) {
      v8bf lo = *(const v8bf*)(ar + 32 * ks);
      v8bf hi = *(const v8bf*)(ar + 32 * ks + 8);
      A[t][ks] = __builtin_shufflevector(lo, hi, 0, 1, 2, 3, 4, 5, 6, 7,
                                         8, 9, 10, 11, 12, 13, 14, 15);
    }
  }

#pragma unroll
  for (int nt = 0; nt < 8; ++nt) {
    const int n = nt * 16 + m;
    const float b = bias[n];
    v8f acc0 = {b, b, b, b, b, b, b, b};
    v8f acc1 = acc0;
    const bf16* br = sW + n * DD + kofB;
#pragma unroll
    for (int ks = 0; ks < 4; ++ks) {
      v16bf B = *(const v16bf*)(br + 32 * ks);
      acc0 = wmma_bf16(A[0][ks], B, acc0);
      acc1 = wmma_bf16(A[1][ks], B, acc1);
    }
#pragma unroll
    for (int t = 0; t < 2; ++t) {
      v8f a = t ? acc1 : acc0;
#pragma unroll
      for (int r = 0; r < 8; ++r) {
        int row = rowBase + 16 * t + rHalf + r;
        if (WB == 0) {
          sXw[row * DD + n] = (bf16)gelu_fast(a[r]);
        } else if (WB == 1) {
          if (row < rows)
            __hip_atomic_fetch_add(gout + (size_t)dsts[8 * t + r] * DD + n,
                                   a[r], __ATOMIC_RELAXED,
                                   __HIP_MEMORY_SCOPE_AGENT);
        } else {
          if (row < rows)
            gout[(size_t)(gRowBase + row) * DD + n] = a[r];
        }
      }
    }
  }
}

// ---------------------------------------------------------------------------
__global__ void zero_f4(float4* __restrict__ p, int n4) {
  int i = blockIdx.x * blockDim.x + threadIdx.x;
  if (i < n4) p[i] = make_float4(0.f, 0.f, 0.f, 0.f);
}

// ---------------------------------------------------------------------------
__global__ __launch_bounds__(NTHREADS)
void gine_edge_kernel(const float* __restrict__ x,
                      const int* __restrict__ ei,        // [2*E]: row0=src row1=dst
                      const float* __restrict__ ea,      // [E, D]
                      const float* __restrict__ W1, const float* __restrict__ b1,
                      const float* __restrict__ W2, const float* __restrict__ b2,
                      float* __restrict__ agg, int E_) {
  __shared__ alignas(32) bf16 sW[DD * DD];   // 32 KB: W1^T then W2^T
  __shared__ alignas(32) bf16 sX[BR * DD];   // 32 KB: input rows then hidden

  const int e0   = blockIdx.x * BR;
  const int rows = (E_ - e0 < BR) ? (E_ - e0) : BR;
  const int lane = threadIdx.x & 31;
  const int wave = threadIdx.x >> 5;
  const int rowBase = wave * 32;
  const int rHalf   = (lane & 16) ? 8 : 0;

  // Gather x[src] + edge_attr, convert to bf16, stage row-major into LDS.
  for (int i = threadIdx.x; i < BR * (DD / 4); i += NTHREADS) {
    int r  = i >> 5;           // DD/4 == 32 float4 per row
    int c4 = i & 31;
    float f0 = 0.f, f1 = 0.f, f2 = 0.f, f3 = 0.f;
    if (r < rows) {
      int e = e0 + r;
      int s = ei[e];
      const float4 a = ((const float4*)(x + (size_t)s * DD))[c4];
      const float4 b = ((const float4*)(ea + (size_t)e * DD))[c4];
      f0 = a.x + b.x; f1 = a.y + b.y; f2 = a.z + b.z; f3 = a.w + b.w;
    }
    v4bf h;
    h[0] = (bf16)f0; h[1] = (bf16)f1; h[2] = (bf16)f2; h[3] = (bf16)f3;
    *(v4bf*)(sX + r * DD + c4 * 4) = h;
  }
  stage_w_t(W1, sW);

  // Hoist this lane's 16 destination node ids into registers (L2-cached).
  int dsts[16];
#pragma unroll
  for (int t = 0; t < 2; ++t)
#pragma unroll
    for (int r = 0; r < 8; ++r) {
      int row = rowBase + 16 * t + rHalf + r;
      dsts[8 * t + r] = (row < rows) ? ei[E_ + e0 + row] : 0;
    }

  __syncthreads();
  run_layer<0>(sX, sX, sW, b1, lane, rowBase, rows, nullptr, nullptr, 0);
  __syncthreads();          // all waves done with W1
  stage_w_t(W2, sW);
  __syncthreads();
  run_layer<1>(sX, nullptr, sW, b2, lane, rowBase, rows, agg, dsts, 0);
}

// ---------------------------------------------------------------------------
__global__ __launch_bounds__(NTHREADS)
void gine_node_kernel(const float* __restrict__ x,
                      const float* __restrict__ agg,
                      const float* __restrict__ W1, const float* __restrict__ b1,
                      const float* __restrict__ W2, const float* __restrict__ b2,
                      const float* __restrict__ eps,
                      float* __restrict__ out, int N_) {
  __shared__ alignas(32) bf16 sW[DD * DD];
  __shared__ alignas(32) bf16 sX[BR * DD];

  const int n0   = blockIdx.x * BR;
  const int rows = (N_ - n0 < BR) ? (N_ - n0) : BR;
  const int lane = threadIdx.x & 31;
  const int wave = threadIdx.x >> 5;
  const int rowBase = wave * 32;
  const float scale = 1.0f + eps[0];

  for (int i = threadIdx.x; i < BR * (DD / 4); i += NTHREADS) {
    int r  = i >> 5;
    int c4 = i & 31;
    float f0 = 0.f, f1 = 0.f, f2 = 0.f, f3 = 0.f;
    if (r < rows) {
      size_t base = (size_t)(n0 + r) * DD;
      const float4 a = ((const float4*)(x + base))[c4];
      const float4 g = ((const float4*)(agg + base))[c4];
      f0 = scale * a.x + g.x; f1 = scale * a.y + g.y;
      f2 = scale * a.z + g.z; f3 = scale * a.w + g.w;
    }
    v4bf h;
    h[0] = (bf16)f0; h[1] = (bf16)f1; h[2] = (bf16)f2; h[3] = (bf16)f3;
    *(v4bf*)(sX + r * DD + c4 * 4) = h;
  }
  stage_w_t(W1, sW);
  __syncthreads();

  run_layer<0>(sX, sX, sW, b1, lane, rowBase, rows, nullptr, nullptr, 0);
  __syncthreads();
  stage_w_t(W2, sW);
  __syncthreads();
  run_layer<2>(sX, nullptr, sW, b2, lane, rowBase, rows, out, nullptr, n0);
}

// ---------------------------------------------------------------------------
extern "C" void kernel_launch(void* const* d_in, const int* in_sizes, int n_in,
                              void* d_out, int out_size, void* d_ws, size_t ws_size,
                              hipStream_t stream) {
  const float* x   = (const float*)d_in[0];
  const int*   ei  = (const int*)  d_in[1];   // [2, E]
  const float* ea  = (const float*)d_in[2];
  const float* We1 = (const float*)d_in[3];
  const float* be1 = (const float*)d_in[4];
  const float* We2 = (const float*)d_in[5];
  const float* be2 = (const float*)d_in[6];
  const float* Wu1 = (const float*)d_in[7];
  const float* bu1 = (const float*)d_in[8];
  const float* Wu2 = (const float*)d_in[9];
  const float* bu2 = (const float*)d_in[10];
  const float* eps = (const float*)d_in[11];

  float* agg = (float*)d_ws;        // [N, D] f32 aggregation buffer
  float* out = (float*)d_out;

  // Zero the aggregation buffer every call (graph replays, atomics accumulate).
  {
    int n4 = NNODES * DD / 4;
    int blocks = (n4 + 255) / 256;
    zero_f4<<<blocks, 256, 0, stream>>>((float4*)agg, n4);
  }

  {
    int blocks = (NEDGES + BR - 1) / BR;   // 4688
    gine_edge_kernel<<<blocks, NTHREADS, 0, stream>>>(
        x, ei, ea, We1, be1, We2, be2, agg, NEDGES);
  }

  {
    int blocks = (NNODES + BR - 1) / BR;   // 391
    gine_node_kernel<<<blocks, NTHREADS, 0, stream>>>(
        x, agg, Wu1, bu1, Wu2, bu2, eps, out, NNODES);
  }
}
